// ExtraPositionPromptSABottleneck_51204600103022
// MI455X (gfx1250) — matrix operand order repaired and checked by reading, not compiled, not run on hardware
//
#include <hip/hip_runtime.h>

#define BATCH 8
#define DIMS 512
#define CH   256
#define NPIX 4096   // 64*64
#define SZ   64

typedef unsigned int u32;
typedef __attribute__((ext_vector_type(16))) __bf16 v16bf;
typedef __attribute__((ext_vector_type(8)))  __bf16 v8bf;
typedef __attribute__((ext_vector_type(4)))  __bf16 v4bf;
typedef __attribute__((ext_vector_type(8)))  float  v8f;
typedef __attribute__((ext_vector_type(4)))  u32    v4u32;
typedef __attribute__((ext_vector_type(8)))  int    v8i32;
typedef __attribute__((ext_vector_type(4)))  int    v4i32;

#define WMMA_BF16(A_, B_, C_) \
  __builtin_amdgcn_wmma_f32_16x16x32_bf16(false, (A_), false, (B_), (short)0, (C_), false, false)

// A fragment: 16(M)x32(K) bf16 from LDS [M][ldk].
__device__ __forceinline__ v16bf frag_a(const __bf16* lds, int ldk, int mBase, int k0, int lane) {
  const __bf16* p = lds + (size_t)(mBase + (lane & 15)) * ldk + k0 + ((lane >> 4) << 4);
  v8bf lo = *(const v8bf*)(p);
  v8bf hi = *(const v8bf*)(p + 8);
  v16bf r;
#pragma unroll
  for (int i = 0; i < 8; ++i) { r[i] = lo[i]; r[i + 8] = hi[i]; }
  return r;
}

// B fragment: 32(K)x16(N) bf16 from LDS staged transposed [N][ldk] (32B contiguous per lane).
__device__ __forceinline__ v16bf frag_b(const __bf16* lds, int ldk, int nBase, int k0, int lane) {
  return *(const v16bf*)(lds + (size_t)(nBase + (lane & 15)) * ldk + k0 + ((lane >> 4) << 4));
}

__device__ __forceinline__ float silu_f(float x) { return x / (1.f + __expf(-x)); }

// ---------------------------------------------------------------------------
// Tensor Data Mover: 2D tile (global, 2-byte elems) -> LDS.
// 6-arg builtin flavor: (v4u32 g0, v8i32 g1, v4i32, v4i32, v8i32, i32 cpol).
// ---------------------------------------------------------------------------
__device__ __forceinline__ void tdm_load_2d(u32 lds_off, const __bf16* gsrc,
                                            u32 tensor_w, u32 tile_w, u32 tile_h,
                                            u32 row_stride,
                                            u32 pad_en, u32 pad_int, u32 pad_amt) {
  unsigned long long ga = (unsigned long long)(size_t)(const void*)gsrc;
  v4u32 g0;
  g0[0] = 1u;                                          // count=1, user descriptor
  g0[1] = lds_off;                                     // lds_addr (bytes)
  g0[2] = (u32)ga;                                     // global_addr[31:0]
  g0[3] = ((u32)(ga >> 32) & 0x1FFFFFFu) | (2u << 30); // addr[56:32] | type=2
  v8i32 g1;
  g1[0] = (int)((1u << 16) | (pad_en << 20) | (pad_int << 22) | (pad_amt << 25)); // data_size=2B
  g1[1] = (int)((tensor_w & 0xFFFFu) << 16);           // tensor_dim0[15:0]
  g1[2] = (int)((tensor_w >> 16) | ((tile_h & 0xFFFFu) << 16)); // dim0 hi | tensor_dim1 lo
  g1[3] = (int)((tile_h >> 16) | (tile_w << 16));      // tensor_dim1 hi | tile_dim0
  g1[4] = (int)(tile_h & 0xFFFFu);                     // tile_dim1 (tile_dim2=0)
  g1[5] = (int)row_stride;                             // tensor_dim0_stride[31:0]
  g1[6] = 0;
  g1[7] = 0;
  v4i32 z4 = {0, 0, 0, 0};
  v8i32 z8 = {0, 0, 0, 0, 0, 0, 0, 0};
  __builtin_amdgcn_tensor_load_to_lds(g0, g1, z4, z4, z8, 0);
}

__device__ __forceinline__ u32 lds_off_of(const void* p) {
  return (u32)(size_t)p;   // LDS aperture: addr[31:0] is the LDS offset
}

// ---------------------------------------------------------------------------
// Bulk f32 -> bf16 convert (weights; n multiple of 4). One-time prep.
// ---------------------------------------------------------------------------
__global__ __launch_bounds__(256) void k_cvt(const float* __restrict__ s,
                                             __bf16* __restrict__ d) {
  int i = blockIdx.x * 256 + threadIdx.x;
  float4 f = ((const float4*)s)[i];
  v4bf o; o[0] = (__bf16)f.x; o[1] = (__bf16)f.y; o[2] = (__bf16)f.z; o[3] = (__bf16)f.w;
  ((v4bf*)d)[i] = o;
}

// ---------------------------------------------------------------------------
// pos_t[n][c] = rel_h[c][h] + rel_w[c][w]  (bf16, transposed, batch-invariant)
// ---------------------------------------------------------------------------
__global__ __launch_bounds__(256) void k_pos(const float* __restrict__ rel_h,
                                             const float* __restrict__ rel_w,
                                             __bf16* __restrict__ pos_t) {
  int idx = blockIdx.x * 256 + threadIdx.x;     // idx = n*CH + c
  int c = idx & (CH - 1), n = idx >> 8;
  int h = n >> 6, w = n & 63;
  pos_t[idx] = (__bf16)(rel_h[c * SZ + h] + rel_w[c * SZ + w]);
}

// ---------------------------------------------------------------------------
// x1_t[b][n][c] = silu(bn1(W1 @ x)). A tile via TDM (bf16 weights);
// B tile needs f32->bf16 transpose-convert (manual).
// ---------------------------------------------------------------------------
__global__ __launch_bounds__(128) void k_cv1(
    const float* __restrict__ x, const __bf16* __restrict__ w1b,
    const float* __restrict__ g, const float* __restrict__ bta,
    const float* __restrict__ mu, const float* __restrict__ var,
    __bf16* __restrict__ x1t) {
  __shared__ __bf16 As[64 * 32];   // [m][k]
  __shared__ __bf16 Bs[64 * 32];   // [n][k]
  const int b = blockIdx.z;
  const int n0 = blockIdx.x * 64;
  const int m0 = blockIdx.y * 64;
  const int tid = threadIdx.x, lane = tid & 31, wv = tid >> 5;
  v8f acc[4] = {};
  for (int k0 = 0; k0 < DIMS; k0 += 32) {
    __syncthreads();
    if (wv == 0)
      tdm_load_2d(lds_off_of(As), w1b + (size_t)m0 * DIMS + k0, 32u, 32u, 64u, DIMS, 0u, 0u, 0u);
#pragma unroll
    for (int i = 0; i < 4; ++i) {               // B: x[b][k][n] column reads -> Bs[n][k] packed
      int lin = tid + i * 128;
      int nn = lin & 63, kk = (lin >> 6) << 2;
      const float* src = x + ((size_t)b * DIMS + k0 + kk) * NPIX + n0 + nn;
      v4bf o;
#pragma unroll
      for (int j = 0; j < 4; ++j) o[j] = (__bf16)src[(size_t)j * NPIX];
      *(v4bf*)(&Bs[nn * 32 + kk]) = o;
    }
    if (wv == 0) __builtin_amdgcn_s_wait_tensorcnt((short)0);
    __syncthreads();
    v16bf a = frag_a(As, 32, wv * 16, 0, lane);
#pragma unroll
    for (int nt = 0; nt < 4; ++nt) {
      v16bf bb = frag_b(Bs, 32, nt * 16, 0, lane);
      acc[nt] = WMMA_BF16(a, bb, acc[nt]);
    }
  }
  const int mB = m0 + wv * 16 + ((lane >> 4) << 3);   // channel base (contiguous 8)
  const int nB = n0 + (lane & 15);
  float sc[8], sh[8];
#pragma unroll
  for (int r = 0; r < 8; ++r) {
    int c = mB + r;
    sc[r] = g[c] * rsqrtf(var[c] + 1e-5f);
    sh[r] = bta[c] - mu[c] * sc[r];
  }
#pragma unroll
  for (int nt = 0; nt < 4; ++nt) {
    v8bf o;
#pragma unroll
    for (int r = 0; r < 8; ++r) o[r] = (__bf16)silu_f(acc[nt][r] * sc[r] + sh[r]);
    *(v8bf*)(x1t + ((size_t)b * NPIX + nB + nt * 16) * CH + mB) = o;   // b128 store
  }
}

// ---------------------------------------------------------------------------
// Projections (all-bf16 GEMM; both LDS tiles staged by TDM).
// q -> Qt[b][n][256], k/e -> K2t[b][n][512], v -> Vb[b][c][n].
// ---------------------------------------------------------------------------
__global__ __launch_bounds__(128) void k_qkve(
    const __bf16* __restrict__ x1t,
    const __bf16* __restrict__ wqb, const __bf16* __restrict__ wkb,
    const __bf16* __restrict__ wvb, const __bf16* __restrict__ web,
    const float* __restrict__ qb, const float* __restrict__ kb,
    const float* __restrict__ vb, const float* __restrict__ eb,
    __bf16* __restrict__ Qt, __bf16* __restrict__ K2t, __bf16* __restrict__ Vb) {
  __shared__ __bf16 As[64 * 32];
  __shared__ __bf16 Bs[64 * 32];
  const int b = blockIdx.z;
  const int n0 = blockIdx.x * 64;
  const int m0 = blockIdx.y * 64;
  const int quarter = m0 >> 8;
  const int mloc = m0 & 255;
  const int tid = threadIdx.x, lane = tid & 31, wv = tid >> 5;
  const __bf16* wp; const float* bp;
  if (quarter == 0)      { wp = wqb; bp = qb; }
  else if (quarter == 1) { wp = wkb; bp = kb; }
  else if (quarter == 2) { wp = wvb; bp = vb; }
  else                   { wp = web; bp = eb; }
  v8f acc[4] = {};
  for (int k0 = 0; k0 < CH; k0 += 32) {
    __syncthreads();
    if (wv == 0) {
      tdm_load_2d(lds_off_of(As), wp + (size_t)mloc * CH + k0, 32u, 32u, 64u, CH, 0u, 0u, 0u);
      tdm_load_2d(lds_off_of(Bs), x1t + ((size_t)b * NPIX + n0) * CH + k0, 32u, 32u, 64u, CH,
                  0u, 0u, 0u);
      __builtin_amdgcn_s_wait_tensorcnt((short)0);
    }
    __syncthreads();
    v16bf a = frag_a(As, 32, wv * 16, 0, lane);
#pragma unroll
    for (int nt = 0; nt < 4; ++nt) {
      v16bf bb = frag_b(Bs, 32, nt * 16, 0, lane);
      acc[nt] = WMMA_BF16(a, bb, acc[nt]);
    }
  }
  const int rr0 = wv * 16 + ((lane >> 4) << 3);
  const int nB = n0 + (lane & 15);
  float bias[8];
#pragma unroll
  for (int r = 0; r < 8; ++r) bias[r] = bp[mloc + rr0 + r];
  if (quarter == 2) {                            // v: natural [c][n]
#pragma unroll
    for (int r = 0; r < 8; ++r)
#pragma unroll
      for (int nt = 0; nt < 4; ++nt)
        Vb[((size_t)b * CH + mloc + rr0 + r) * NPIX + nB + nt * 16] =
            (__bf16)(acc[nt][r] + bias[r]);
  } else {                                       // q/k/e: transposed, contiguous b128 stores
    __bf16* base = (quarter == 0) ? Qt : K2t;
    const int ld = (quarter == 0) ? CH : 2 * CH;
    const int col0 = ((quarter == 3) ? CH : 0) + mloc + rr0;
#pragma unroll
    for (int nt = 0; nt < 4; ++nt) {
      v8bf o;
#pragma unroll
      for (int r = 0; r < 8; ++r) o[r] = (__bf16)(acc[nt][r] + bias[r]);
      *(v8bf*)(base + ((size_t)b * NPIX + nB + nt * 16) * ld + col0) = o;
    }
  }
}

// ---------------------------------------------------------------------------
// Flash attention with double-buffered TDM staging: iteration i+1's K'/V DMAs
// overlap iteration i's 384 WMMAs; wave 0 waits TENSORcnt<=2 (cur done, next
// in flight). Q' resident in LDS; all staging is Tensor Data Mover.
// ---------------------------------------------------------------------------
__global__ __launch_bounds__(256) void k_attn(
    const __bf16* __restrict__ Qt, const __bf16* __restrict__ pos_t,
    const __bf16* __restrict__ K2t, const __bf16* __restrict__ Vb,
    __bf16* __restrict__ Obt) {
  __shared__ __bf16 Qs[64 * 512];      // [query][512ch]   64 KB
  __shared__ __bf16 Ks[2][64 * 512];   // [key][512ch]    128 KB (double buffer)
  __shared__ __bf16 Vs[2][256 * 64];   // [ch][key]        64 KB (double buffer)
  __shared__ __bf16 Ps[64 * 64];       // [query][key]      8 KB
  __shared__ float  Ss[64 * 64];       // [query][key]     16 KB
  __shared__ float  red[256];
  __shared__ float  rmS[64], rlS[64], alph[64];
  const int b = blockIdx.y;
  const int q0 = blockIdx.x * 64;
  const int tid = threadIdx.x, lane = tid & 31, wv = tid >> 5;

  auto issue_kv = [&](int kbase, int p) {
    // K' block: 64 rows x 512ch contiguous in K2t -> one flat 64KB DMA.
    tdm_load_2d(lds_off_of(&Ks[p][0]), K2t + ((size_t)b * NPIX + kbase) * (2 * CH),
                32768u, 32768u, 1u, 32768u, 0u, 0u, 0u);
    // V block: 256 rows x 64 keys, row stride NPIX.
    tdm_load_2d(lds_off_of(&Vs[p][0]), Vb + (size_t)b * CH * NPIX + kbase,
                NPIX, 64u, CH, NPIX, 0u, 0u, 0u);
  };

  if (wv == 0) {
    // Q' block: q channels -> cols 0..255, pos -> 256..511 of each 1KB LDS row.
    // pad: after every 128 DWORDs (512B, code 6) insert 128 DWORDs (512B, code 127).
    tdm_load_2d(lds_off_of(Qs),       Qt    + ((size_t)b * NPIX + q0) * CH,
                CH, CH, 64, CH, 1u, 6u, 127u);
    tdm_load_2d(lds_off_of(Qs) + 512, pos_t + (size_t)q0 * CH,
                CH, CH, 64, CH, 1u, 6u, 127u);
    issue_kv(0, 0);
  }
  if (tid < 64) { rmS[tid] = -3.4e38f; rlS[tid] = 0.f; }
  v8f acc[2][4] = {};                            // wave owns channels [wv*32, wv*32+32)

  const int NBLK = NPIX / 64;
  for (int ib = 0; ib < NBLK; ++ib) {
    const int p = ib & 1;
    __syncthreads();                             // all readers of buffer p^1 done
    if (wv == 0) {
      if (ib + 1 < NBLK) {
        issue_kv((ib + 1) * 64, p ^ 1);          // prefetch next block
        __builtin_amdgcn_s_wait_tensorcnt((short)2);  // only next 2 still in flight
      } else {
        __builtin_amdgcn_s_wait_tensorcnt((short)0);
      }
    }
    __syncthreads();                             // current buffer visible to all waves
    // S = Q'^T K' over K=512: wave does tiles (qt, kt0) and (qt, kt0+1): shared A-frag.
    {
      const int qt = wv >> 1, kt0 = (wv & 1) << 1;
      v8f s0 = {}, s1 = {};
      const __bf16* KsC = &Ks[p][0];
#pragma unroll 4
      for (int kk = 0; kk < 512; kk += 32) {
        v16bf a  = frag_a(Qs, 512, qt * 16, kk, lane);
        v16bf b0 = frag_b(KsC, 512, kt0 * 16, kk, lane);
        v16bf b1 = frag_b(KsC, 512, kt0 * 16 + 16, kk, lane);
        s0 = WMMA_BF16(a, b0, s0);
        s1 = WMMA_BF16(a, b1, s1);
      }
      int qr = qt * 16 + ((lane >> 4) << 3);
      int kc = kt0 * 16 + (lane & 15);
#pragma unroll
      for (int r = 0; r < 8; ++r) {
        Ss[(qr + r) * 64 + kc] = s0[r];
        Ss[(qr + r) * 64 + kc + 16] = s1[r];
      }
    }
    __syncthreads();
    // Online softmax, 4 threads per query row.
    {
      int row = tid >> 2, sub = tid & 3;
      const float* sr = Ss + row * 64 + sub * 16;
      float lm = -3.4e38f;
#pragma unroll
      for (int j = 0; j < 16; ++j) lm = fmaxf(lm, sr[j]);
      red[tid] = lm;
    }
    __syncthreads();
    if (tid < 64) {
      float mx = fmaxf(fmaxf(red[tid * 4], red[tid * 4 + 1]),
                       fmaxf(red[tid * 4 + 2], red[tid * 4 + 3]));
      float old = rmS[tid];
      mx = fmaxf(old, mx);
      alph[tid] = __expf(old - mx);
      rmS[tid] = mx;
    }
    __syncthreads();
    {
      int row = tid >> 2, sub = tid & 3;
      float mx = rmS[row];
      const float* sr = Ss + row * 64 + sub * 16;
      __bf16* pr = Ps + row * 64 + sub * 16;
      float s = 0.f;
#pragma unroll
      for (int j = 0; j < 16; ++j) {
        float pj = __expf(sr[j] - mx);
        s += pj;
        pr[j] = (__bf16)pj;
      }
      red[tid] = s;
    }
    __syncthreads();                             // Ps complete; partial sums ready
    if (tid < 64)
      rlS[tid] = rlS[tid] * alph[tid] +
                 (red[tid * 4] + red[tid * 4 + 1] + red[tid * 4 + 2] + red[tid * 4 + 3]);
    // Rescale running accumulators (each lane's D-column = one fixed query).
#pragma unroll
    for (int nt = 0; nt < 4; ++nt) {
      float al = alph[nt * 16 + (lane & 15)];
#pragma unroll
      for (int mt = 0; mt < 2; ++mt)
#pragma unroll
        for (int r = 0; r < 8; ++r) acc[mt][nt][r] *= al;
    }
    // O += V * P^T  (V A-frags hoisted: 6 frag loads per kk for 8 WMMAs)
    {
      const __bf16* VsC = &Vs[p][0];
#pragma unroll
      for (int kk = 0; kk < 64; kk += 32) {
        v16bf a0 = frag_a(VsC, 64, wv * 32, kk, lane);
        v16bf a1 = frag_a(VsC, 64, wv * 32 + 16, kk, lane);
#pragma unroll
        for (int nt = 0; nt < 4; ++nt) {
          v16bf bb = frag_b(Ps, 64, nt * 16, kk, lane);
          acc[0][nt] = WMMA_BF16(a0, bb, acc[0][nt]);
          acc[1][nt] = WMMA_BF16(a1, bb, acc[1][nt]);
        }
      }
    }
  }
  __syncthreads();                               // rlS final
#pragma unroll
  for (int nt = 0; nt < 4; ++nt) {
    float li = 1.f / rlS[nt * 16 + (lane & 15)];
    int qc = q0 + nt * 16 + (lane & 15);
#pragma unroll
    for (int mt = 0; mt < 2; ++mt) {
      int c0 = wv * 32 + mt * 16 + ((lane >> 4) << 3);
      v8bf o;
#pragma unroll
      for (int r = 0; r < 8; ++r) o[r] = (__bf16)(acc[mt][nt][r] * li);
      *(v8bf*)(Obt + ((size_t)b * NPIX + qc) * CH + c0) = o;   // b128 store
    }
  }
}

// ---------------------------------------------------------------------------
// out = silu(bn2(W2 @ O)) + x  (fp32, natural layout; both tiles via TDM)
// ---------------------------------------------------------------------------
__global__ __launch_bounds__(128) void k_cv2(
    const __bf16* __restrict__ Obt, const __bf16* __restrict__ w2b,
    const float* __restrict__ g, const float* __restrict__ bta,
    const float* __restrict__ mu, const float* __restrict__ var,
    const float* __restrict__ x, float* __restrict__ out) {
  __shared__ __bf16 As[64 * 32];
  __shared__ __bf16 Bs[64 * 32];
  const int b = blockIdx.z;
  const int n0 = blockIdx.x * 64;
  const int m0 = blockIdx.y * 64;
  const int tid = threadIdx.x, lane = tid & 31, wv = tid >> 5;
  v8f acc[4] = {};
  for (int k0 = 0; k0 < CH; k0 += 32) {
    __syncthreads();
    if (wv == 0) {
      tdm_load_2d(lds_off_of(As), w2b + (size_t)m0 * CH + k0, 32u, 32u, 64u, CH, 0u, 0u, 0u);
      tdm_load_2d(lds_off_of(Bs), Obt + ((size_t)b * NPIX + n0) * CH + k0, 32u, 32u, 64u, CH,
                  0u, 0u, 0u);
      __builtin_amdgcn_s_wait_tensorcnt((short)0);
    }
    __syncthreads();
    v16bf a = frag_a(As, 32, wv * 16, 0, lane);
#pragma unroll
    for (int nt = 0; nt < 4; ++nt) {
      v16bf bb = frag_b(Bs, 32, nt * 16, 0, lane);
      acc[nt] = WMMA_BF16(a, bb, acc[nt]);
    }
  }
  const int mB = m0 + wv * 16 + ((lane >> 4) << 3);
  const int nB = n0 + (lane & 15);
#pragma unroll
  for (int r = 0; r < 8; ++r) {
    int d = mB + r;
    float sc = g[d] * rsqrtf(var[d] + 1e-5f);
    float sh = bta[d] - mu[d] * sc;
#pragma unroll
    for (int nt = 0; nt < 4; ++nt) {
      size_t idx = ((size_t)b * DIMS + d) * NPIX + nB + nt * 16;
      float v = acc[nt][r] * sc + sh;
      out[idx] = silu_f(v) + x[idx];
    }
  }
}

extern "C" void kernel_launch(void* const* d_in, const int* in_sizes, int n_in,
                              void* d_out, int out_size, void* d_ws, size_t ws_size,
                              hipStream_t stream) {
  (void)in_sizes; (void)n_in; (void)out_size; (void)ws_size;
  const float* x     = (const float*)d_in[0];
  const float* cv1_w = (const float*)d_in[1];
  const float* bn1_g = (const float*)d_in[2];
  const float* bn1_b = (const float*)d_in[3];
  const float* bn1_m = (const float*)d_in[4];
  const float* bn1_v = (const float*)d_in[5];
  const float* q_w   = (const float*)d_in[6];
  const float* q_b   = (const float*)d_in[7];
  const float* k_w   = (const float*)d_in[8];
  const float* k_b   = (const float*)d_in[9];
  const float* v_w   = (const float*)d_in[10];
  const float* v_b   = (const float*)d_in[11];
  const float* e_w   = (const float*)d_in[12];
  const float* e_b   = (const float*)d_in[13];
  const float* rel_h = (const float*)d_in[14];
  const float* rel_w = (const float*)d_in[15];
  const float* cv2_w = (const float*)d_in[16];
  const float* bn2_g = (const float*)d_in[17];
  const float* bn2_b = (const float*)d_in[18];
  const float* bn2_m = (const float*)d_in[19];
  const float* bn2_v = (const float*)d_in[20];
  float* out = (float*)d_out;

  char* ws = (char*)d_ws;
  __bf16* pos_t = (__bf16*)ws;  ws += (size_t)NPIX * CH * 2;              //  2 MB  [n][c]
  __bf16* x1t   = (__bf16*)ws;  ws += (size_t)BATCH * NPIX * CH * 2;      // 16 MB  [b][n][c]
  __bf16* Qt    = (__bf16*)ws;  ws += (size_t)BATCH * NPIX * CH * 2;      // 16 MB  [b][n][c]
  __bf16* K2t   = (__bf16*)ws;  ws += (size_t)BATCH * NPIX * 2 * CH * 2;  // 32 MB  [b][n][2c]
  __bf16* Vb    = (__bf16*)ws;  ws += (size_t)BATCH * CH * NPIX * 2;      // 16 MB  [b][c][n]
  __bf16* Obt   = (__bf16*)ws;  ws += (size_t)BATCH * NPIX * CH * 2;      // 16 MB  [b][n][c]
  __bf16* w1b   = (__bf16*)ws;  ws += (size_t)CH * DIMS * 2;              // bf16 weights
  __bf16* wqb   = (__bf16*)ws;  ws += (size_t)CH * CH * 2;
  __bf16* wkb   = (__bf16*)ws;  ws += (size_t)CH * CH * 2;
  __bf16* wvb   = (__bf16*)ws;  ws += (size_t)CH * CH * 2;
  __bf16* web   = (__bf16*)ws;  ws += (size_t)CH * CH * 2;
  __bf16* w2b   = (__bf16*)ws;

  // One-time weight conversion to bf16 (vectorized, n/4 threads).
  k_cvt<<<dim3(CH * DIMS / 1024), dim3(256), 0, stream>>>(cv1_w, w1b);
  k_cvt<<<dim3(CH * CH / 1024), dim3(256), 0, stream>>>(q_w, wqb);
  k_cvt<<<dim3(CH * CH / 1024), dim3(256), 0, stream>>>(k_w, wkb);
  k_cvt<<<dim3(CH * CH / 1024), dim3(256), 0, stream>>>(v_w, wvb);
  k_cvt<<<dim3(CH * CH / 1024), dim3(256), 0, stream>>>(e_w, web);
  k_cvt<<<dim3(DIMS * CH / 1024), dim3(256), 0, stream>>>(cv2_w, w2b);
  k_pos <<<dim3(CH * NPIX / 256), dim3(256), 0, stream>>>(rel_h, rel_w, pos_t);

  k_cv1 <<<dim3(NPIX / 64, CH / 64, BATCH), dim3(128), 0, stream>>>(
      x, w1b, bn1_g, bn1_b, bn1_m, bn1_v, x1t);
  k_qkve<<<dim3(NPIX / 64, (4 * CH) / 64, BATCH), dim3(128), 0, stream>>>(
      x1t, wqb, wkb, wvb, web, q_b, k_b, v_b, e_b, Qt, K2t, Vb);
  k_attn<<<dim3(NPIX / 64, BATCH), dim3(256), 0, stream>>>(Qt, pos_t, K2t, Vb, Obt);
  k_cv2 <<<dim3(NPIX / 64, DIMS / 64, BATCH), dim3(128), 0, stream>>>(
      Obt, w2b, bn2_g, bn2_b, bn2_m, bn2_v, x, out);
}
